// GraphEncoder_66099546686020
// MI455X (gfx1250) — matrix operand order repaired and checked by reading, not compiled, not run on hardware
//
#include <hip/hip_runtime.h>

// ---------------- problem constants (match reference) ----------------
#define NN   16384
#define BB   32
#define SS   512
#define EE   262144
#define DD   256
#define LL   4
#define HH   4
#define DHH  64
#define OUTD 768

typedef __attribute__((ext_vector_type(16))) __bf16         bf16x16;
typedef __attribute__((ext_vector_type(8)))  float          floatx8;
typedef __attribute__((ext_vector_type(8)))  unsigned short u16x8;
typedef __attribute__((ext_vector_type(16))) unsigned short u16x16;

// -------- helpers --------
__device__ __forceinline__ unsigned short f2bf(float f) {
    unsigned int u = __builtin_bit_cast(unsigned int, f);
    u = (u + 0x7FFFu + ((u >> 16) & 1u)) >> 16;
    return (unsigned short)u;
}
// A-fragment: elements 0..7 at p[0..7], 8..15 at p[16..23]  (two b128 loads)
__device__ __forceinline__ bf16x16 load_gap(const unsigned short* p) {
    u16x8 lo = *(const u16x8*)p;
    u16x8 hi = *(const u16x8*)(p + 16);
    u16x16 c = __builtin_shufflevector(lo, hi, 0,1,2,3,4,5,6,7,8,9,10,11,12,13,14,15);
    return __builtin_bit_cast(bf16x16, c);
}
// B-fragment: elements 0..7 at p[0..7], 8..15 at p[16..23]  (two b128 loads)
__device__ __forceinline__ bf16x16 load_contig(const unsigned short* p) {
    u16x8 lo = *(const u16x8*)p;
    u16x8 hi = *(const u16x8*)(p + 16);
    u16x16 c = __builtin_shufflevector(lo, hi, 0,1,2,3,4,5,6,7,8,9,10,11,12,13,14,15);
    return __builtin_bit_cast(bf16x16, c);
}
// CDNA5 async memory->LDS copy (16B per lane), tracked by ASYNCcnt.
__device__ __forceinline__ void async_copy_b128(unsigned lds_off, const void* gptr) {
    asm volatile("global_load_async_to_lds_b128 %0, %1, off"
                 :: "v"(lds_off), "v"(gptr) : "memory");
}
__device__ __forceinline__ void wait_asynccnt0() {
    asm volatile("s_wait_asynccnt 0" ::: "memory");
}

// =====================================================================
// One-time weight prep: f32 [L][K][M] -> bf16 transposed [L][M][K]
// =====================================================================
__global__ void wcvt_kernel(const float* __restrict__ W, unsigned short* __restrict__ out,
                            int K, int M, int total) {
    int idx = blockIdx.x * blockDim.x + threadIdx.x;
    if (idx >= total) return;
    int km = K * M;
    int l = idx / km, rem = idx - l * km;
    int k = rem / M, m = rem - k * M;
    out[(size_t)l * km + (size_t)m * K + k] = f2bf(W[idx]);
}

// =====================================================================
// Encoders
// =====================================================================
__global__ void atom_encode_kernel(const int* __restrict__ x, const float* __restrict__ tab,
                                   float* __restrict__ h, unsigned short* __restrict__ hbf) {
    int idx = blockIdx.x * blockDim.x + threadIdx.x;
    if (idx >= NN * DD) return;
    int n = idx >> 8, d = idx & 255;
    float s = 0.f;
#pragma unroll
    for (int f = 0; f < 9; ++f) {
        int v = x[n * 9 + f];
        s += tab[((size_t)f * 119 + v) * DD + d];
    }
    h[idx] = s;
    hbf[idx] = f2bf(s);
}

__global__ void bond_encode_kernel(const int* __restrict__ ea, const float* __restrict__ tab,
                                   unsigned short* __restrict__ ebf) {
    int idx = blockIdx.x * blockDim.x + threadIdx.x;
    if (idx >= EE * DD) return;
    int n = idx >> 8, d = idx & 255;
    float s = 0.f;
#pragma unroll
    for (int f = 0; f < 3; ++f) {
        int v = ea[n * 3 + f];
        s += tab[((size_t)f * 22 + v) * DD + d];
    }
    ebf[idx] = f2bf(s);
}

// =====================================================================
// Generic bf16 WMMA GEMM: out = [relu](A[N,K] @ Wt[M,K]^T + bias) (+res)
// 2D grid: blockIdx.y = column tile; 8 waves per block over row tiles.
// =====================================================================
template <bool RELU, bool HAS_RES, bool WRITE_F, bool WRITE_B>
__global__ void gemm_bf16_kernel(const unsigned short* __restrict__ A,
                                 const unsigned short* __restrict__ Wt,
                                 const float* __restrict__ bias,
                                 const float* __restrict__ res,
                                 float* __restrict__ outf,
                                 unsigned short* __restrict__ outb,
                                 int K, int M) {
    int lane  = threadIdx.x & 31;
    int wave  = threadIdx.x >> 5;
    int n16   = lane & 15;
    int khalf = lane >> 4;
    int rt = blockIdx.x * 8 + wave;                      // row tile (N/16 divisible by 8)
    int ct = blockIdx.y;                                 // column tile

    floatx8 acc = {};
    const unsigned short* arow = A + (size_t)(rt * 16 + n16) * K;
    const unsigned short* brow = Wt + (size_t)(ct * 16 + n16) * K;
    int ksteps = K >> 5;
    for (int ks = 0; ks < ksteps; ++ks) {
        bf16x16 af = load_gap(arow + ks * 32 + khalf * 8);
        bf16x16 bf = load_contig(brow + ks * 32 + khalf * 8);
        acc = __builtin_amdgcn_wmma_f32_16x16x32_bf16(false, af, false, bf,
                                                      (short)0, acc, false, false);
    }
    int gc = ct * 16 + n16;
    float bz = bias[gc];
#pragma unroll
    for (int r = 0; r < 8; ++r) {
        int gr = rt * 16 + r + khalf * 8;                // C/D layout
        size_t o = (size_t)gr * M + gc;
        float v = acc[r] + bz;
        if (HAS_RES) v += res[o];
        if (RELU) v = fmaxf(v, 0.f);
        if (WRITE_F) outf[o] = v;
        if (WRITE_B) outb[o] = f2bf(v);
    }
}

// =====================================================================
// Fused GINE edge path: el = e@W+b ; msg = relu(h[src]+el) ; agg[dst] += msg
// 16 waves/block, one col-tile per wave, weights register-resident
// (8 B-fragments = 64 VGPRs). A-tile (8KB) staged via ASYNC memory->LDS.
// =====================================================================
__global__ void edge_msg_kernel(const unsigned short* __restrict__ Ebf,
                                const unsigned short* __restrict__ Wt,   // [256][256] bf16
                                const float* __restrict__ bias,
                                const float* __restrict__ h,
                                const int* __restrict__ src,
                                const int* __restrict__ dst,
                                float* __restrict__ agg,
                                int numTiles) {
    __shared__ unsigned short atile[16 * DD];            // 8KB
    __shared__ int sIdx[16], dIdx[16];

    int tid   = threadIdx.x;                             // 512 threads = 16 waves
    int lane  = tid & 31;
    int wave  = tid >> 5;
    int n16   = lane & 15;
    int khalf = lane >> 4;
    int ct    = wave;                                    // 16 column tiles of D=256

    // register-resident B fragments for this wave's 16 output columns
    bf16x16 Breg[8];
    const unsigned short* brow = Wt + (size_t)(ct * 16 + n16) * DD;
#pragma unroll
    for (int ks = 0; ks < 8; ++ks)
        Breg[ks] = load_contig(brow + ks * 32 + khalf * 8);

    int gc = ct * 16 + n16;
    float bz = bias[gc];

    // per-thread staging addresses: 16 rows x 256 bf16 = 4096 u16 / 512 thr = 8 each
    const int i0 = tid * 8;
    const int r0 = i0 >> 8, c0 = i0 & 255;
    const unsigned ldsoff = (unsigned)(size_t)(atile + i0);   // LDS byte offset (low 32b)

    for (int tile = blockIdx.x; tile < numTiles; tile += gridDim.x) {
        __syncthreads();                                 // protect atile reuse
        {
            // async memory -> LDS copy of this thread's 16-byte chunk
            async_copy_b128(ldsoff, Ebf + (size_t)(tile * 16 + r0) * DD + c0);
            if (tid < 16) { sIdx[tid] = src[tile * 16 + tid]; dIdx[tid] = dst[tile * 16 + tid]; }
            wait_asynccnt0();                            // LDS writes landed
        }
        __syncthreads();

        floatx8 acc = {};
#pragma unroll
        for (int ks = 0; ks < 8; ++ks) {
            bf16x16 af = load_gap(atile + n16 * DD + ks * 32 + khalf * 8);
            acc = __builtin_amdgcn_wmma_f32_16x16x32_bf16(false, af, false, Breg[ks],
                                                          (short)0, acc, false, false);
        }
#pragma unroll
        for (int r = 0; r < 8; ++r) {
            int rr = r + khalf * 8;
            int s0 = sIdx[rr], d0 = dIdx[rr];
            float v = acc[r] + bz + h[(size_t)s0 * DD + gc];
            v = fmaxf(v, 0.f);
            atomicAdd(&agg[(size_t)d0 * DD + gc], v);
        }
    }
}

// =====================================================================
// V transpose: qkv[N,768] V-section -> vT[B,H,DH,S] bf16
// =====================================================================
__global__ void vtrans_kernel(const unsigned short* __restrict__ qkv,
                              unsigned short* __restrict__ vT) {
    int idx = blockIdx.x * blockDim.x + threadIdx.x;
    if (idx >= NN * DD) return;
    int n = idx >> 8, d = idx & 255;
    int b = n / SS, s = n - b * SS;
    int hh = d >> 6, dh = d & 63;
    vT[((size_t)(b * HH + hh) * DHH + dh) * SS + s] = qkv[(size_t)n * 768 + 512 + d];
}

// =====================================================================
// Flash-style attention: one wave per (graph b, head hh, 16-query tile).
// =====================================================================
__global__ void attn_kernel(const unsigned short* __restrict__ qkv,
                            const unsigned short* __restrict__ vT,
                            unsigned short* __restrict__ obf) {
    __shared__ float          sc[16 * SS];               // 32KB
    __shared__ unsigned short pp[16 * SS];               // 16KB

    int qt = blockIdx.x & 31;
    int hh = (blockIdx.x >> 5) & 3;
    int bq = blockIdx.x >> 7;

    int lane  = threadIdx.x & 31;
    int n16   = lane & 15;
    int khalf = lane >> 4;
    const float scale = 0.125f;                          // 1/sqrt(64)

    // ---- scores S = Q K^T ----
    const unsigned short* qbase =
        qkv + (size_t)(bq * SS + qt * 16 + n16) * 768 + hh * DHH;
    for (int kt = 0; kt < 32; ++kt) {
        const unsigned short* kbase =
            qkv + (size_t)(bq * SS + kt * 16 + n16) * 768 + 256 + hh * DHH;
        floatx8 acc = {};
#pragma unroll
        for (int ks = 0; ks < 2; ++ks) {                 // DH=64 -> 2 k-steps of 32
            bf16x16 af = load_gap(qbase + ks * 32 + khalf * 8);
            bf16x16 bf = load_contig(kbase + ks * 32 + khalf * 8);
            acc = __builtin_amdgcn_wmma_f32_16x16x32_bf16(false, af, false, bf,
                                                          (short)0, acc, false, false);
        }
#pragma unroll
        for (int r = 0; r < 8; ++r)
            sc[(r + khalf * 8) * SS + kt * 16 + n16] = acc[r] * scale;
    }
    __syncthreads();

    // ---- softmax over 512 keys (row = n16, lane halves split columns) ----
    {
        const int base = n16 * SS + khalf * 256;
        float mx = -3.4e38f;
        for (int c = 0; c < 256; ++c) mx = fmaxf(mx, sc[base + c]);
        mx = fmaxf(mx, __shfl_xor(mx, 16));
        float sum = 0.f;
        for (int c = 0; c < 256; ++c) {
            float ev = __expf(sc[base + c] - mx);
            sc[base + c] = ev;
            sum += ev;
        }
        sum += __shfl_xor(sum, 16);
        float inv = 1.f / sum;
        for (int c0 = 0; c0 < 256; c0 += 8) {
            u16x8 v;
#pragma unroll
            for (int i = 0; i < 8; ++i) v[i] = f2bf(sc[base + c0 + i] * inv);
            *(u16x8*)(pp + base + c0) = v;
        }
    }
    __syncthreads();

    // ---- O = P @ V  (16x512 @ 512x64), V from transposed [DH][S] layout ----
    const unsigned short* prow = pp + n16 * SS;
    for (int ctv = 0; ctv < 4; ++ctv) {
        const unsigned short* vrow =
            vT + ((size_t)(bq * HH + hh) * DHH + ctv * 16 + n16) * SS;
        floatx8 acc = {};
        for (int ks = 0; ks < 16; ++ks) {
            bf16x16 af = load_gap(prow + ks * 32 + khalf * 8);
            bf16x16 bf = load_contig(vrow + ks * 32 + khalf * 8);
            acc = __builtin_amdgcn_wmma_f32_16x16x32_bf16(false, af, false, bf,
                                                          (short)0, acc, false, false);
        }
#pragma unroll
        for (int r = 0; r < 8; ++r) {
            int gr = bq * SS + qt * 16 + r + khalf * 8;
            obf[(size_t)gr * DD + hh * DHH + ctv * 16 + n16] = f2bf(acc[r]);
        }
    }
}

// =====================================================================
// Elementwise / BN / pooling / head kernels
// =====================================================================
__global__ void gine_combine_kernel(const float* __restrict__ h, const float* __restrict__ agg,
                                    const float* __restrict__ eps, int l,
                                    float* __restrict__ y, unsigned short* __restrict__ ybf) {
    int idx = blockIdx.x * blockDim.x + threadIdx.x;
    if (idx >= NN * DD) return;
    float v = (1.f + eps[l]) * h[idx] + agg[idx];
    y[idx] = v;
    ybf[idx] = f2bf(v);
}

__global__ void add2_kernel(const float* __restrict__ a, const float* __restrict__ b,
                            float* __restrict__ y, unsigned short* __restrict__ ybf) {
    int idx = blockIdx.x * blockDim.x + threadIdx.x;
    if (idx >= NN * DD) return;
    float v = a[idx] + b[idx];
    y[idx] = v;
    ybf[idx] = f2bf(v);
}

__global__ void colstats_kernel(const float* __restrict__ x, float* __restrict__ mean,
                                float* __restrict__ rstd) {
    __shared__ float rs[256], rss[256];
    int d = blockIdx.x, tid = threadIdx.x;
    float s = 0.f, ss = 0.f;
    for (int n = tid; n < NN; n += 256) {
        float v = x[(size_t)n * DD + d];
        s += v; ss += v * v;
    }
    rs[tid] = s; rss[tid] = ss;
    __syncthreads();
    for (int off = 128; off > 0; off >>= 1) {
        if (tid < off) { rs[tid] += rs[tid + off]; rss[tid] += rss[tid + off]; }
        __syncthreads();
    }
    if (tid == 0) {
        float mu = rs[0] / NN;
        float var = rss[0] / NN - mu * mu;
        mean[d] = mu;
        rstd[d] = rsqrtf(var + 1e-5f);
    }
}

__global__ void bn_apply_kernel(const float* __restrict__ x, const float* __restrict__ mean,
                                const float* __restrict__ rstd, const float* __restrict__ g,
                                const float* __restrict__ b, float* __restrict__ y,
                                unsigned short* __restrict__ ybf) {
    int idx = blockIdx.x * blockDim.x + threadIdx.x;
    if (idx >= NN * DD) return;
    int d = idx & 255;
    float v = (x[idx] - mean[d]) * rstd[d] * g[d] + b[d];
    y[idx] = v;
    if (ybf) ybf[idx] = f2bf(v);
}

__global__ void pool_kernel(const float* __restrict__ h, float* __restrict__ g) {
    int b = blockIdx.x, d = threadIdx.x;
    float s = 0.f;
    for (int sdx = 0; sdx < SS; ++sdx) s += h[(size_t)(b * SS + sdx) * DD + d];
    g[b * DD + d] = s * (1.f / (float)SS);
}

__global__ void head1_kernel(const float* __restrict__ g, const float* __restrict__ pw1,
                             const float* __restrict__ pb1, float* __restrict__ t) {
    int b = blockIdx.x, j = threadIdx.x;
    float s = pb1[j];
    for (int d = 0; d < DD; ++d) s += g[b * DD + d] * pw1[d * DD + j];
    t[b * DD + j] = fmaxf(s, 0.f);
}

__global__ void head2_kernel(const float* __restrict__ t, const float* __restrict__ pw2,
                             const float* __restrict__ pb2, float* __restrict__ z) {
    int idx = blockIdx.x * blockDim.x + threadIdx.x;
    if (idx >= BB * OUTD) return;
    int b = idx / OUTD, o = idx - b * OUTD;
    float s = pb2[o];
    for (int j = 0; j < DD; ++j) s += t[b * DD + j] * pw2[j * OUTD + o];
    z[idx] = s;
}

__global__ void l2norm_kernel(float* __restrict__ z) {
    __shared__ float red[256];
    int b = blockIdx.x, tid = threadIdx.x;
    float s = 0.f;
    for (int o = tid; o < OUTD; o += 256) { float v = z[b * OUTD + o]; s += v * v; }
    red[tid] = s;
    __syncthreads();
    for (int off = 128; off > 0; off >>= 1) {
        if (tid < off) red[tid] += red[tid + off];
        __syncthreads();
    }
    float inv = rsqrtf(red[0]);
    for (int o = tid; o < OUTD; o += 256) z[b * OUTD + o] *= inv;
}

// =====================================================================
// Host-side orchestration
// =====================================================================
extern "C" void kernel_launch(void* const* d_in, const int* in_sizes, int n_in,
                              void* d_out, int out_size, void* d_ws, size_t ws_size,
                              hipStream_t stream) {
    (void)in_sizes; (void)n_in; (void)out_size; (void)ws_size;

    const int*   x        = (const int*)d_in[0];
    const int*   eattr    = (const int*)d_in[1];
    const int*   eidx     = (const int*)d_in[2];
    const float* atom_tab = (const float*)d_in[4];
    const float* bond_tab = (const float*)d_in[5];
    const float* elin_w   = (const float*)d_in[6];
    const float* elin_b   = (const float*)d_in[7];
    const float* eps      = (const float*)d_in[8];
    const float* gw1 = (const float*)d_in[9],  *gb1 = (const float*)d_in[10];
    const float* gw2 = (const float*)d_in[11], *gb2 = (const float*)d_in[12];
    const float* aw_in  = (const float*)d_in[13], *ab_in  = (const float*)d_in[14];
    const float* aw_out = (const float*)d_in[15], *ab_out = (const float*)d_in[16];
    const float* n1g = (const float*)d_in[17], *n1b = (const float*)d_in[18];
    const float* n2g = (const float*)d_in[19], *n2b = (const float*)d_in[20];
    const float* n3g = (const float*)d_in[21], *n3b = (const float*)d_in[22];
    const float* mw1 = (const float*)d_in[23], *mb1 = (const float*)d_in[24];
    const float* mw2 = (const float*)d_in[25], *mb2 = (const float*)d_in[26];
    const float* pw1 = (const float*)d_in[27], *pb1 = (const float*)d_in[28];
    const float* pw2 = (const float*)d_in[29], *pb2 = (const float*)d_in[30];

    const int* src = eidx;
    const int* dst = eidx + EE;

    // -------- workspace layout --------
    size_t off = 0;
    auto alloc = [&](size_t bytes) -> char* {
        char* p = (char*)d_ws + off;
        off += (bytes + 255) & ~(size_t)255;
        return p;
    };
    float*          h      = (float*)alloc((size_t)NN * DD * 4);
    unsigned short* h_bf   = (unsigned short*)alloc((size_t)NN * DD * 2);
    unsigned short* e_bf   = (unsigned short*)alloc((size_t)EE * DD * 2);
    float*          agg    = (float*)alloc((size_t)NN * DD * 4);
    float*          x1     = (float*)alloc((size_t)NN * DD * 4);
    unsigned short* x1_bf  = (unsigned short*)alloc((size_t)NN * DD * 2);
    unsigned short* t2x    = (unsigned short*)alloc((size_t)NN * 2 * DD * 2);
    float*          pre    = (float*)alloc((size_t)NN * DD * 4);
    float*          hc     = (float*)alloc((size_t)NN * DD * 4);
    float*          ha     = (float*)alloc((size_t)NN * DD * 4);
    unsigned short* qkv_bf = (unsigned short*)alloc((size_t)NN * 768 * 2);
    unsigned short* vT     = (unsigned short*)alloc((size_t)NN * DD * 2);
    unsigned short* o_bf   = (unsigned short*)alloc((size_t)NN * DD * 2);
    unsigned short* elinT  = (unsigned short*)alloc((size_t)LL * DD * DD * 2);
    unsigned short* gw1T   = (unsigned short*)alloc((size_t)LL * DD * 2 * DD * 2);
    unsigned short* gw2T   = (unsigned short*)alloc((size_t)LL * 2 * DD * DD * 2);
    unsigned short* awinT  = (unsigned short*)alloc((size_t)LL * DD * 3 * DD * 2);
    unsigned short* awoutT = (unsigned short*)alloc((size_t)LL * DD * DD * 2);
    unsigned short* mw1T   = (unsigned short*)alloc((size_t)LL * DD * 2 * DD * 2);
    unsigned short* mw2T   = (unsigned short*)alloc((size_t)LL * 2 * DD * DD * 2);
    float*          meanB  = (float*)alloc(DD * 4);
    float*          rstdB  = (float*)alloc(DD * 4);
    float*          gpool  = (float*)alloc(BB * DD * 4);
    float*          thead  = (float*)alloc(BB * DD * 4);
    float*          zout   = (float*)d_out;

    const int ELEM_BLOCKS = (NN * DD) / 256;             // 16384
    const int ROWB = (NN / 16) / 8;                      // 128 row-tile blocks

    auto wcvt = [&](const float* W, unsigned short* out, int K, int M) {
        int total = LL * K * M;
        wcvt_kernel<<<(total + 255) / 256, 256, 0, stream>>>(W, out, K, M, total);
    };
    auto batchnorm = [&](const float* xin, const float* gg, const float* bb,
                         float* y, unsigned short* ybf) {
        colstats_kernel<<<DD, 256, 0, stream>>>(xin, meanB, rstdB);
        bn_apply_kernel<<<ELEM_BLOCKS, 256, 0, stream>>>(xin, meanB, rstdB, gg, bb, y, ybf);
    };

    // -------- one-time weight conversion (bf16, transposed [M][K]) --------
    wcvt(elin_w, elinT,  DD,     DD);
    wcvt(gw1,    gw1T,   DD,     2 * DD);
    wcvt(gw2,    gw2T,   2 * DD, DD);
    wcvt(aw_in,  awinT,  DD,     3 * DD);
    wcvt(aw_out, awoutT, DD,     DD);
    wcvt(mw1,    mw1T,   DD,     2 * DD);
    wcvt(mw2,    mw2T,   2 * DD, DD);

    // -------- encoders --------
    atom_encode_kernel<<<ELEM_BLOCKS, 256, 0, stream>>>(x, atom_tab, h, h_bf);
    bond_encode_kernel<<<(EE * DD) / 256, 256, 0, stream>>>(eattr, bond_tab, e_bf);

    // -------- layers --------
    for (int l = 0; l < LL; ++l) {
        // GINE local branch
        hipMemsetAsync(agg, 0, (size_t)NN * DD * 4, stream);
        edge_msg_kernel<<<1024, 512, 0, stream>>>(
            e_bf, elinT + (size_t)l * DD * DD, elin_b + l * DD, h, src, dst, agg, EE / 16);
        gine_combine_kernel<<<ELEM_BLOCKS, 256, 0, stream>>>(h, agg, eps, l, x1, x1_bf);
        gemm_bf16_kernel<true, false, false, true>
            <<<dim3(ROWB, (2 * DD) / 16), 256, 0, stream>>>(
            x1_bf, gw1T + (size_t)l * DD * 2 * DD, gb1 + l * 2 * DD,
            nullptr, nullptr, t2x, DD, 2 * DD);
        gemm_bf16_kernel<false, true, true, false>
            <<<dim3(ROWB, DD / 16), 256, 0, stream>>>(
            t2x, gw2T + (size_t)l * 2 * DD * DD, gb2 + l * DD,
            h, pre, nullptr, 2 * DD, DD);
        batchnorm(pre, n1g + l * DD, n1b + l * DD, hc, nullptr);

        // global attention branch
        gemm_bf16_kernel<false, false, false, true>
            <<<dim3(ROWB, (3 * DD) / 16), 256, 0, stream>>>(
            h_bf, awinT + (size_t)l * DD * 3 * DD, ab_in + l * 3 * DD,
            nullptr, nullptr, qkv_bf, DD, 3 * DD);
        vtrans_kernel<<<ELEM_BLOCKS, 256, 0, stream>>>(qkv_bf, vT);
        attn_kernel<<<BB * HH * (SS / 16), 32, 0, stream>>>(qkv_bf, vT, o_bf);
        gemm_bf16_kernel<false, true, true, false>
            <<<dim3(ROWB, DD / 16), 256, 0, stream>>>(
            o_bf, awoutT + (size_t)l * DD * DD, ab_out + l * DD,
            h, pre, nullptr, DD, DD);
        batchnorm(pre, n2g + l * DD, n2b + l * DD, ha, nullptr);

        // combine + FFN
        add2_kernel<<<ELEM_BLOCKS, 256, 0, stream>>>(hc, ha, x1, x1_bf);
        gemm_bf16_kernel<true, false, false, true>
            <<<dim3(ROWB, (2 * DD) / 16), 256, 0, stream>>>(
            x1_bf, mw1T + (size_t)l * DD * 2 * DD, mb1 + l * 2 * DD,
            nullptr, nullptr, t2x, DD, 2 * DD);
        gemm_bf16_kernel<false, true, true, false>
            <<<dim3(ROWB, DD / 16), 256, 0, stream>>>(
            t2x, mw2T + (size_t)l * 2 * DD * DD, mb2 + l * DD,
            x1, pre, nullptr, 2 * DD, DD);
        batchnorm(pre, n3g + l * DD, n3b + l * DD, h, h_bf);
    }

    // -------- pool + head + normalize --------
    pool_kernel<<<BB, 256, 0, stream>>>(h, gpool);
    head1_kernel<<<BB, 256, 0, stream>>>(gpool, pw1, pb1, thead);
    head2_kernel<<<(BB * OUTD + 255) / 256, 256, 0, stream>>>(thead, pw2, pb2, zout);
    l2norm_kernel<<<BB, 256, 0, stream>>>(zout);
}
